// SelfAttentionWithRelPos_52149492908597
// MI455X (gfx1250) — matrix operand order repaired and checked by reading, not compiled, not run on hardware
//
#include <hip/hip_runtime.h>
#include <hip/hip_bf16.h>

// Problem constants (match reference)
#define B_   2
#define S_   2048
#define D_   1024
#define H_   16
#define HD_  64
// NUM_BUCKETS=32, MAX_DISTANCE=128 -> bucket = n<16 ? n : min(31, 16 + int(log(n/16)/log(8)*16))

typedef __attribute__((ext_vector_type(16))) _Float16     v16h;
typedef __attribute__((ext_vector_type(8)))  _Float16     v8h;
typedef __attribute__((ext_vector_type(8)))  float        v8f;
typedef __attribute__((ext_vector_type(4)))  unsigned int u32x4_t;
typedef __attribute__((ext_vector_type(8)))  int          i32x8_t;
typedef __attribute__((ext_vector_type(4)))  int          i32x4_t;

union AFrag { v16h v; v8h h[2]; };

// ---- WMMA helpers (CDNA5 wave32 layouts, ISA 7.12.2) --------------------
// A 16x32 f16: lane&15 = M row; hi=lane>>4: elems 0..7 -> K=hi*8..+7,
//                                          elems 8..15 -> K=16+hi*8..+7
__device__ __forceinline__ v16h load_a_frag_g(const _Float16* __restrict__ base,
                                              int ldk, int row0, int k0, int lane) {
  const int r  = row0 + (lane & 15);
  const int hi = lane >> 4;
  const _Float16* p = base + (size_t)r * ldk + (k0 + hi * 8);
  AFrag f;
  f.h[0] = *reinterpret_cast<const v8h*>(p);
  f.h[1] = *reinterpret_cast<const v8h*>(p + 16);
  return f.v;
}

// B 32x16 f16 (B[k][n] = Wrow[n][k]): lane&15 = N col; elems 0..7 -> K=k0+hi*8..+7, elems 8..15 -> K=k0+16+hi*8..+7
__device__ __forceinline__ v16h load_b_frag_g(const _Float16* __restrict__ base,
                                              int ldk, int col0, int k0, int lane) {
  const int n  = col0 + (lane & 15);
  const int kb = k0 + (lane >> 4) * 8;
  const _Float16* p = base + (size_t)n * ldk + kb;
  AFrag f;
  f.h[0] = *reinterpret_cast<const v8h*>(p);
  f.h[1] = *reinterpret_cast<const v8h*>(p + 16);
  return f.v;
}

// B fragment out of an LDS-resident tile: rows = N (32 f16 of K each)
__device__ __forceinline__ v16h load_b_frag_lds(const _Float16* base, int col0,
                                                int lane) {
  const int n  = col0 + (lane & 15);
  const int kb = (lane >> 4) * 8;
  const _Float16* p = base + n * 32 + kb;
  AFrag f;
  f.h[0] = *reinterpret_cast<const v8h*>(p);
  f.h[1] = *reinterpret_cast<const v8h*>(p + 16);
  return f.v;
}

__device__ __forceinline__ v8f wmma_f16(v16h a, v16h b, v8f c) {
  return __builtin_amdgcn_wmma_f32_16x16x32_f16(false, a, false, b, (short)0, c,
                                                false, false);
}

// ---- Tensor Data Mover: 2-D f16 tile (tile_rows x tile_k) -> LDS --------
// D# packing per CDNA5 ISA ch.8: group0 = count/lds/global/type,
// group1 = data_size + tensor/tile dims + dim0 stride. Groups 2/3 = 0 (2-D).
// This toolchain exposes the 6-arg builtin (extra int32x8 group before cpol).
__device__ __forceinline__ void tdm_load_tile_2d(const _Float16* gptr,
                                                 unsigned lds_off, int tile_k,
                                                 int tile_rows, int row_stride) {
  const unsigned long long ga = (unsigned long long)(size_t)gptr;
  u32x4_t g0;
  g0[0] = 1u;                                         // count=1, user desc
  g0[1] = lds_off;                                    // lds_addr (bytes)
  g0[2] = (unsigned)(ga & 0xffffffffu);               // global_addr[31:0]
  g0[3] = (unsigned)((ga >> 32) & 0x01ffffffu) | (2u << 30);  // [56:32] | type=2
  i32x8_t g1;
  g1[0] = (1 << 16);                                  // data_size=1 (2B elems)
  g1[1] = (tile_k & 0xffff) << 16;                    // tensor_dim0[15:0]
  g1[2] = ((tile_k >> 16) & 0xffff) | ((tile_rows & 0xffff) << 16);  // dim0 hi | dim1 lo
  g1[3] = ((tile_rows >> 16) & 0xffff) | ((tile_k & 0xffff) << 16);  // dim1 hi | tile_dim0
  g1[4] = (tile_rows & 0xffff);                       // tile_dim1 (tile_dim2=0)
  g1[5] = row_stride;                                 // tensor_dim0_stride[31:0]
  g1[6] = 0;
  g1[7] = 0;
  const i32x4_t z4 = {0, 0, 0, 0};
  const i32x8_t z8 = {0, 0, 0, 0, 0, 0, 0, 0};
  __builtin_amdgcn_tensor_load_to_lds(g0, g1, z4, z4, z8, 0);
}

// ---- fp32 -> f16 cast ----------------------------------------------------
__global__ void cast_f32_to_f16(const float* __restrict__ src,
                                _Float16* __restrict__ dst, size_t n) {
  size_t i = (size_t)blockIdx.x * blockDim.x + threadIdx.x;
  size_t stride = (size_t)gridDim.x * blockDim.x;
  for (; i < n; i += stride) dst[i] = (_Float16)src[i];
}

// ---- Y[M,N] = X[M,K] @ W[N,K]^T ------------------------------------------
// Block = 256 thr (8 waves), tile 128x128. W K-slabs (128x32 f16, 8 KB) are
// staged into LDS by the Tensor Data Mover, double-buffered, overlapping the
// WMMA stream. Each wave computes 32x64 = 8 WMMAs per K step.
template <bool F16OUT>
__global__ void __launch_bounds__(256)
gemm_xwT(const _Float16* __restrict__ X, const _Float16* __restrict__ W,
         void* __restrict__ Yv, int M, int N, int K) {
  __shared__ __align__(16) _Float16 Wtile[2][128 * 32];   // 16 KB double buffer

  const int lane = threadIdx.x & 31;
  const int wv   = threadIdx.x >> 5;
  const int wm   = wv >> 1;                 // 0..3 -> M sub-tile
  const int wn   = wv & 1;                  // 0..1 -> N sub-tile
  const int m0   = blockIdx.y * 128 + wm * 32;
  const int nblk = blockIdx.x * 128;
  const int n0l  = wn * 64;                 // local N base inside LDS tile

  const _Float16* wbase = W + (size_t)nblk * K;

  // prologue: TDM-stage first K slab
  if (wv == 0)
    tdm_load_tile_2d(wbase, (unsigned)(size_t)(const void*)&Wtile[0][0], 32, 128, K);

  v8f acc[2][4];
#pragma unroll
  for (int mi = 0; mi < 2; ++mi)
#pragma unroll
    for (int ni = 0; ni < 4; ++ni) acc[mi][ni] = v8f{};

  for (int k0 = 0; k0 < K; k0 += 32) {
    const int cur = (k0 >> 5) & 1;
    if (wv == 0) __builtin_amdgcn_s_wait_tensorcnt(0);  // slab `cur` landed
    __syncthreads();                                    // publish to all waves
    if (wv == 0 && (k0 + 32) < K)                       // async-stage next slab
      tdm_load_tile_2d(wbase + (k0 + 32),
                       (unsigned)(size_t)(const void*)&Wtile[cur ^ 1][0],
                       32, 128, K);

    v16h a0 = load_a_frag_g(X, K, m0,      k0, lane);
    v16h a1 = load_a_frag_g(X, K, m0 + 16, k0, lane);
    if (k0 + 32 < K) {  // prefetch next A slab (global_prefetch_b8)
      __builtin_prefetch(X + (size_t)(m0 + (lane & 15)) * K + k0 + 32, 0, 1);
      __builtin_prefetch(X + (size_t)(m0 + 16 + (lane & 15)) * K + k0 + 32, 0, 1);
    }
    const _Float16* bt = &Wtile[cur][0];
    v16h b0 = load_b_frag_lds(bt, n0l,      lane);
    v16h b1 = load_b_frag_lds(bt, n0l + 16, lane);
    v16h b2 = load_b_frag_lds(bt, n0l + 32, lane);
    v16h b3 = load_b_frag_lds(bt, n0l + 48, lane);

    acc[0][0] = wmma_f16(a0, b0, acc[0][0]);
    acc[0][1] = wmma_f16(a0, b1, acc[0][1]);
    acc[0][2] = wmma_f16(a0, b2, acc[0][2]);
    acc[0][3] = wmma_f16(a0, b3, acc[0][3]);
    acc[1][0] = wmma_f16(a1, b0, acc[1][0]);
    acc[1][1] = wmma_f16(a1, b1, acc[1][1]);
    acc[1][2] = wmma_f16(a1, b2, acc[1][2]);
    acc[1][3] = wmma_f16(a1, b3, acc[1][3]);
  }

  const int hi = lane >> 4;
  const int ln = lane & 15;
#pragma unroll
  for (int mi = 0; mi < 2; ++mi) {
#pragma unroll
    for (int ni = 0; ni < 4; ++ni) {
      const int cn = nblk + n0l + ni * 16 + ln;
#pragma unroll
      for (int r = 0; r < 8; ++r) {
        const int rm = m0 + mi * 16 + r + 8 * hi;
        if (F16OUT) {
          ((_Float16*)Yv)[(size_t)rm * N + cn] = (_Float16)acc[mi][ni][r];
        } else {
          ((float*)Yv)[(size_t)rm * N + cn] = acc[mi][ni][r];
        }
      }
    }
  }
}

// ---- V (B*S, D) row-major -> Vt (B,H,HD,S) so PV K-dim is contiguous -----
__global__ void transpose_v(const _Float16* __restrict__ Vr,
                            _Float16* __restrict__ Vt, size_t n) {
  size_t i = (size_t)blockIdx.x * blockDim.x + threadIdx.x;
  if (i >= n) return;
  const int    d  = (int)(i % D_);
  const size_t sg = i / D_;
  const int    s  = (int)(sg % S_);
  const int    b  = (int)(sg / S_);
  const int    h  = d >> 6;
  const int    dd = d & 63;
  Vt[((size_t)(b * H_ + h) * HD_ + dd) * S_ + s] = Vr[i];
}

// ---- fused flash attention: 1 wave = 16 query rows, 32-key tiles ---------
__global__ void __launch_bounds__(128)
attention_kernel(const _Float16* __restrict__ Qh, const _Float16* __restrict__ Kh,
                 const _Float16* __restrict__ Vt, const float* __restrict__ rel_emb,
                 const unsigned char* __restrict__ padding_mask,
                 _Float16* __restrict__ Oh) {
  __shared__ __align__(16) float    bias_tab[S_ + 32];   // idx = (q-k)+32
  __shared__ __align__(16) float    pad_tab[S_];
  __shared__ __align__(16) _Float16 Pbuf[4][16 * 32];    // per-wave P tile

  const int tid  = threadIdx.x;
  const int lane = tid & 31;
  const int wv   = tid >> 5;
  const int qt   = blockIdx.x % (S_ / 64);
  const int bh   = blockIdx.x / (S_ / 64);
  const int h    = bh % H_;
  const int b    = bh / H_;
  const int q0   = qt * 64 + wv * 16;

  // Precompute (bias + causal mask) table over delta = q-k, and padding table.
  const float kLogScale = 16.0f / 2.0794415416798357f;   // 16 / ln(8)
  for (int i = tid; i < S_ + 32; i += 128) {
    const int delta = i - 32;
    float v;
    if (delta < 0) {
      v = -1e30f;                                        // causal: k > q
    } else if (delta < 16) {
      v = rel_emb[delta * H_ + h];
    } else {
      int t = 16 + (int)(__logf((float)delta * 0.0625f) * kLogScale);
      t = t < 31 ? t : 31;
      v = rel_emb[t * H_ + h];
    }
    bias_tab[i] = v;
  }
  for (int k = tid; k < S_; k += 128)
    pad_tab[k] = padding_mask[(size_t)b * S_ + k] ? -1e30f : 0.0f;
  __syncthreads();

  const _Float16* qbase = Qh + (size_t)b * S_ * D_ + h * HD_;
  const _Float16* kbase = Kh + (size_t)b * S_ * D_ + h * HD_;
  const _Float16* vbase = Vt + (size_t)(b * H_ + h) * HD_ * S_;

  const v16h qa0 = load_a_frag_g(qbase, D_, q0, 0,  lane);
  const v16h qa1 = load_a_frag_g(qbase, D_, q0, 32, lane);

  v8f o0 = {}, o1 = {}, o2 = {}, o3 = {};
  float mrow[8], lrow[8];
#pragma unroll
  for (int r = 0; r < 8; ++r) { mrow[r] = -3.0e38f; lrow[r] = 0.0f; }

  const int hi = lane >> 4;
  const int ln = lane & 15;
  _Float16* pw = &Pbuf[wv][0];

  for (int kt = 0; kt < q0 + 16; kt += 32) {
    // scores s0: keys kt..kt+15, s1: kt+16..kt+31  (K-dim = HD = 64)
    v8f s0 = {}, s1 = {};
    {
      v16h kb0 = load_b_frag_g(kbase, D_, kt,      0,  lane);
      v16h kb1 = load_b_frag_g(kbase, D_, kt,      32, lane);
      v16h kb2 = load_b_frag_g(kbase, D_, kt + 16, 0,  lane);
      v16h kb3 = load_b_frag_g(kbase, D_, kt + 16, 32, lane);
      s0 = wmma_f16(qa0, kb0, s0);
      s0 = wmma_f16(qa1, kb1, s0);
      s1 = wmma_f16(qa0, kb2, s1);
      s1 = wmma_f16(qa1, kb3, s1);
    }

    const int   c0   = kt + ln;
    const int   c1   = c0 + 16;
    const float pad0 = pad_tab[c0];
    const float pad1 = pad_tab[c1];

#pragma unroll
    for (int r = 0; r < 8; ++r) {
      const int q = q0 + r + 8 * hi;
      float t0 = s0[r] * 0.125f + bias_tab[q - c0 + 32] + pad0;
      float t1 = s1[r] * 0.125f + bias_tab[q - c1 + 32] + pad1;
      // row max over the 16 lanes holding this row's columns
      float tm = fmaxf(t0, t1);
      tm = fmaxf(tm, __shfl_xor(tm, 1));
      tm = fmaxf(tm, __shfl_xor(tm, 2));
      tm = fmaxf(tm, __shfl_xor(tm, 4));
      tm = fmaxf(tm, __shfl_xor(tm, 8));
      const float mnew  = fmaxf(mrow[r], tm);
      const float alpha = __expf(mrow[r] - mnew);
      const float p0    = __expf(t0 - mnew);
      const float p1    = __expf(t1 - mnew);
      float ps = p0 + p1;
      ps += __shfl_xor(ps, 1);
      ps += __shfl_xor(ps, 2);
      ps += __shfl_xor(ps, 4);
      ps += __shfl_xor(ps, 8);
      lrow[r] = lrow[r] * alpha + ps;
      mrow[r] = mnew;
      o0[r] *= alpha; o1[r] *= alpha; o2[r] *= alpha; o3[r] *= alpha;
      // spill P in C-layout to LDS (row-major 16x32)
      const int prow = r + 8 * hi;
      pw[prow * 32 + ln]      = (_Float16)p0;
      pw[prow * 32 + 16 + ln] = (_Float16)p1;
    }

    // reload P in A-layout from LDS (ds_load_b128 x2)
    AFrag pf;
    {
      const _Float16* pp = pw + ln * 32 + hi * 8;
      pf.h[0] = *reinterpret_cast<const v8h*>(pp);
      pf.h[1] = *reinterpret_cast<const v8h*>(pp + 16);
    }

    // O += P @ V  (Vt rows = hd, contiguous along sequence)
    v16h vb0 = load_b_frag_g(vbase, S_, 0,  kt, lane);
    v16h vb1 = load_b_frag_g(vbase, S_, 16, kt, lane);
    v16h vb2 = load_b_frag_g(vbase, S_, 32, kt, lane);
    v16h vb3 = load_b_frag_g(vbase, S_, 48, kt, lane);
    o0 = wmma_f16(pf.v, vb0, o0);
    o1 = wmma_f16(pf.v, vb1, o1);
    o2 = wmma_f16(pf.v, vb2, o2);
    o3 = wmma_f16(pf.v, vb3, o3);
  }

  // normalize and store attention output (f16, token-major)
  _Float16* obase = Oh + (size_t)b * S_ * D_ + h * HD_;
#pragma unroll
  for (int r = 0; r < 8; ++r) {
    const int   row = q0 + r + 8 * hi;
    const float inv = 1.0f / lrow[r];
    _Float16* op = obase + (size_t)row * D_;
    op[ln]      = (_Float16)(o0[r] * inv);
    op[16 + ln] = (_Float16)(o1[r] * inv);
    op[32 + ln] = (_Float16)(o2[r] * inv);
    op[48 + ln] = (_Float16)(o3[r] * inv);
  }
}

// -------------------------------------------------------------------------
extern "C" void kernel_launch(void* const* d_in, const int* in_sizes, int n_in,
                              void* d_out, int out_size, void* d_ws, size_t ws_size,
                              hipStream_t stream) {
  (void)in_sizes; (void)n_in; (void)out_size; (void)ws_size;
  const float*         x   = (const float*)d_in[0];
  const unsigned char* pad = (const unsigned char*)d_in[2];   // padding_mask
  const float*         Wq  = (const float*)d_in[3];
  const float*         Wk  = (const float*)d_in[4];
  const float*         Wv  = (const float*)d_in[5];
  const float*         Wo  = (const float*)d_in[6];
  const float*         rel = (const float*)d_in[7];
  float*               out = (float*)d_out;

  const size_t XN = (size_t)B_ * S_ * D_;   // 4 Mi elems
  const size_t WN = (size_t)D_ * D_;        // 1 Mi elems

  _Float16* ws   = (_Float16*)d_ws;         // 48 MB total
  _Float16* xh   = ws;
  _Float16* wqh  = xh + XN;
  _Float16* wkh  = wqh + WN;
  _Float16* wvh  = wkh + WN;
  _Float16* woh  = wvh + WN;
  _Float16* qh   = woh + WN;
  _Float16* kh   = qh + XN;
  _Float16* vrow = kh + XN;
  _Float16* vt   = vrow + XN;
  _Float16* ah   = xh;  // alias: x dead after the V projection

  const dim3 cb(256);
  cast_f32_to_f16<<<dim3((unsigned)((XN + 255) / 256)), cb, 0, stream>>>(x,  xh,  XN);
  cast_f32_to_f16<<<dim3((unsigned)((WN + 255) / 256)), cb, 0, stream>>>(Wq, wqh, WN);
  cast_f32_to_f16<<<dim3((unsigned)((WN + 255) / 256)), cb, 0, stream>>>(Wk, wkh, WN);
  cast_f32_to_f16<<<dim3((unsigned)((WN + 255) / 256)), cb, 0, stream>>>(Wv, wvh, WN);
  cast_f32_to_f16<<<dim3((unsigned)((WN + 255) / 256)), cb, 0, stream>>>(Wo, woh, WN);

  const int  M = B_ * S_;
  const dim3 gg(D_ / 128, M / 128);
  gemm_xwT<true><<<gg, dim3(256), 0, stream>>>(xh, wqh, qh,   M, D_, D_);
  gemm_xwT<true><<<gg, dim3(256), 0, stream>>>(xh, wkh, kh,   M, D_, D_);
  gemm_xwT<true><<<gg, dim3(256), 0, stream>>>(xh, wvh, vrow, M, D_, D_);

  transpose_v<<<dim3((unsigned)((XN + 255) / 256)), cb, 0, stream>>>(vrow, vt, XN);

  attention_kernel<<<dim3(B_ * H_ * (S_ / 64)), dim3(128), 0, stream>>>(
      qh, kh, vt, rel, pad, ah);

  gemm_xwT<false><<<gg, dim3(256), 0, stream>>>(ah, woh, out, M, D_, D_);
}
